// Net_47571057771091
// MI455X (gfx1250) — matrix operand order repaired and checked by reading, not compiled, not run on hardware
//
#include <hip/hip_runtime.h>
#include <hip/hip_bf16.h>

// Problem constants (from reference): N=2000 nodes, E=4096 edges, H=256.
#define NN 2000
#define EE 4096
#define HH 256
#define KBIG 32768   // H * 128 contraction length for fused edge GEMM
#define KTILES 1024  // KBIG / 32
#define BSTRIDE 16384  // bytes per ktile slab of Bp (16 tiles * 1KB)

typedef __attribute__((ext_vector_type(16))) __bf16 v16bf;
typedef __attribute__((ext_vector_type(8)))  float  v8f;

union FragB  { v16bf v; unsigned int u[8]; };
union FragAcc{ v8f   v; float        f[8]; };

__device__ __forceinline__ unsigned short f2bf(float f) {
  unsigned int u = __float_as_uint(f);
  return (unsigned short)((u + 0x7FFFu + ((u >> 16) & 1u)) >> 16); // RNE
}
__device__ __forceinline__ float bf2f(unsigned short h) {
  return __uint_as_float(((unsigned int)h) << 16);
}
__device__ __forceinline__ unsigned int pack2(float a, float b) {
#if __has_builtin(__builtin_amdgcn_cvt_pk_bf16_f32)
  auto r = __builtin_amdgcn_cvt_pk_bf16_f32(a, b);   // single v_cvt_pk_bf16_f32
  unsigned int u;
  __builtin_memcpy(&u, &r, 4);
  return u;
#else
  return (unsigned int)f2bf(a) | ((unsigned int)f2bf(b) << 16);
#endif
}
__device__ __forceinline__ float sigmf(float x) { return 1.0f / (1.0f + __expf(-x)); }

// ---------------------------------------------------------------------------
// 1) Repack mlp2_W (65536x128 f32) into WMMA-B fragment order, bf16.
//    Bp dword layout: [ktile(1024)][ntile(16)][lane(32)][v(8)]
//    b element (lane,v,j): k = 32*ktile + 16*(lane>>4) + 2*v + j ; n = 16*ntile + (lane&15)
//    global K -> h = K>>7, kk = K&127 ; value = mlp2_W[(h*256+n)*128 + kk]
// ---------------------------------------------------------------------------
__global__ __launch_bounds__(256) void k_pack_B(const float* __restrict__ mlp2_W,
                                                unsigned int* __restrict__ Bp) {
  int tid = blockIdx.x * 256 + threadIdx.x;           // dword index, total 4194304
  int tile = tid >> 8, r = tid & 255;
  int lane = r >> 3, v = r & 7;
  int ktile = tile >> 4, ntile = tile & 15;
  int g = lane >> 4;
  int n = ntile * 16 + (lane & 15);
  int k0 = ktile * 32 + 16 * g + 2 * v;
  int K0 = k0, K1 = k0 + 1;
  float x0 = mlp2_W[((K0 >> 7) * HH + n) * 128 + (K0 & 127)];
  float x1 = mlp2_W[((K1 >> 7) * HH + n) * 128 + (K1 & 127)];
  Bp[tid] = pack2(x0, x1);
}

// ---------------------------------------------------------------------------
// 2) Embedding + lin0 (+ReLU)
// ---------------------------------------------------------------------------
__global__ __launch_bounds__(256) void k_embed_lin0(const int* __restrict__ x,
                                                    const float* __restrict__ pos,
                                                    const float* __restrict__ emb,
                                                    const float* __restrict__ W,
                                                    const float* __restrict__ b,
                                                    float* __restrict__ h0,
                                                    unsigned short* __restrict__ out_bf) {
  __shared__ float s[8];
  int n = blockIdx.x, t = threadIdx.x;
  if (t < 5)            s[t] = emb[x[n] * 5 + t];
  else if (t < 8)       s[t] = pos[n * 3 + (t - 5)];
  __syncthreads();
  float acc = b[t];
#pragma unroll
  for (int j = 0; j < 8; ++j) acc += s[j] * W[t * 8 + j];
  acc = fmaxf(acc, 0.0f);
  h0[n * HH + t] = acc;
  out_bf[n * HH + t] = f2bf(acc);
}

// ---------------------------------------------------------------------------
// 3) Edge feature MLP1 (+ReLU) -> bf16
// ---------------------------------------------------------------------------
__global__ __launch_bounds__(256) void k_he(const float* __restrict__ ea,
                                            const float* __restrict__ W,
                                            const float* __restrict__ b,
                                            unsigned short* __restrict__ he_bf) {
  int tid = blockIdx.x * 256 + threadIdx.x;   // E*128
  int e = tid >> 7, c = tid & 127;
  float acc = b[c];
#pragma unroll
  for (int j = 0; j < 5; ++j) acc += ea[e * 5 + j] * W[c * 5 + j];
  he_bf[tid] = f2bf(fmaxf(acc, 0.0f));
}

// ---------------------------------------------------------------------------
// 4) Fused edge-message GEMM (never materializes W_e):
//    msg[e,o] = sum_K (out_bf[row_e, K>>7] * he_bf[e, K&127]) * Bp[K, o]
//    - he part of the rank-1 A fragment repeats with period 4 in kt:
//      preloaded into 32 registers, zero A-side VMEM in the loop.
//    - A fragment built with one v_pk_mul_bf16 per pair (splat scalar).
//    - B double-buffered; single byte-pointer incremented by BSTRIDE per
//      kstep, loads at immediate offsets -> 1 u64 add per kstep.
// ---------------------------------------------------------------------------
__device__ __forceinline__ void loadB4(const unsigned char* p, FragB b[4]) {
#pragma unroll
  for (int t = 0; t < 4; ++t) {
    uint4 lo = *(const uint4*)(p + t * 1024);
    uint4 hi = *(const uint4*)(p + t * 1024 + 16);
    b[t].u[0] = lo.x; b[t].u[1] = lo.y; b[t].u[2] = lo.z; b[t].u[3] = lo.w;
    b[t].u[4] = hi.x; b[t].u[5] = hi.y; b[t].u[6] = hi.z; b[t].u[7] = hi.w;
  }
}

__global__ __launch_bounds__(256) void k_msg(const unsigned short* __restrict__ out_bf,
                                             const unsigned short* __restrict__ he_bf,
                                             const unsigned int* __restrict__ Bp,
                                             const int* __restrict__ edge_index,
                                             float* __restrict__ msg) {
  const int lane = threadIdx.x & 31;
  const int wave = threadIdx.x >> 5;
  const int eb   = blockIdx.x * 32 + (wave >> 2) * 16;  // 16-edge tile
  const int ncol = (wave & 3) * 64;                     // 64-col slab
  const int g = lane >> 4;
  const int m = lane & 15;
  const int e = eb + m;
  const int node = edge_index[e];                       // row endpoint
  const unsigned short* op = out_bf + (size_t)node * HH;

  // Preload this lane's he dwords: he_reg[kt&3][v] (period-4 reuse pattern)
  unsigned int he_reg[4][8];
  const uint4* hp4 = (const uint4*)(he_bf + (size_t)e * 128);  // 16 uint4 per row
#pragma unroll
  for (int s = 0; s < 4; ++s) {
    uint4 lo = hp4[s * 4 + 2 * g];
    uint4 hi = hp4[s * 4 + 2 * g + 1];
    he_reg[s][0] = lo.x; he_reg[s][1] = lo.y; he_reg[s][2] = lo.z; he_reg[s][3] = lo.w;
    he_reg[s][4] = hi.x; he_reg[s][5] = hi.y; he_reg[s][6] = hi.z; he_reg[s][7] = hi.w;
  }

  FragAcc acc[4];
#pragma unroll
  for (int t = 0; t < 4; ++t)
#pragma unroll
    for (int r = 0; r < 8; ++r) acc[t].f[r] = 0.0f;

  // Per-lane B base (ktile 0); advances by BSTRIDE bytes per kstep.
  const unsigned char* bpb = (const unsigned char*)Bp +
      (((size_t)(ncol >> 4) * 32 + lane) * 8) * 4;

  FragB bb[2][4];
  loadB4(bpb, bb[0]);
  bpb += BSTRIDE;                       // now at kt=1

  for (int hg = 0; hg < 256; ++hg) {    // h index (constant per 4 ksteps)
    unsigned int as = (unsigned int)op[hg];
    unsigned int apk = as * 0x10001u;   // splat bf16 scalar into both halves
    __builtin_prefetch((const void*)(bpb + 7 * BSTRIDE), 0, 1);
#pragma unroll
    for (int sub = 0; sub < 4; ++sub) {
      FragB (&bc)[4] = bb[sub & 1];
      FragB (&bn)[4] = bb[(sub & 1) ^ 1];
      // issue next kstep's B loads (Bp padded by one extra ktile)
      loadB4(bpb, bn);
      bpb += BSTRIDE;
      // A fragment: a[v] = he_reg[sub][v] *pk apk  (packed bf16 multiply)
      FragB a;
#pragma unroll
      for (int v = 0; v < 8; ++v) {
        unsigned int r;
        asm("v_pk_mul_bf16 %0, %1, %2" : "=v"(r) : "v"(he_reg[sub][v]), "v"(apk));
        a.u[v] = r;
      }
#pragma unroll
      for (int t = 0; t < 4; ++t)
        acc[t].v = __builtin_amdgcn_wmma_f32_16x16x32_bf16(
            false, a.v, false, bc[t].v, (short)0, acc[t].v, false, false);
    }
  }

#pragma unroll
  for (int t = 0; t < 4; ++t)
#pragma unroll
    for (int r = 0; r < 8; ++r)
      msg[(size_t)(eb + r + 8 * g) * HH + ncol + t * 16 + m] = acc[t].f[r];
}

// ---------------------------------------------------------------------------
// 5) Generic node GEMM  C[M,N] = A(MxK,f32) @ W(NxK,f32)^T + bias, opt ReLU.
//    bf16 WMMA; K compile-time so the kstep loop fully unrolls and all
//    load offsets become instruction immediates. M multiple of 16.
// ---------------------------------------------------------------------------
template <int KK>
__global__ __launch_bounds__(128) void k_gemm(const float* __restrict__ A,
                                              const float* __restrict__ W,
                                              const float* __restrict__ bias,
                                              float* __restrict__ C,
                                              int Nv, int ldc, int relu) {
  const int lane = threadIdx.x & 31;
  const int wave = threadIdx.x >> 5;
  const int g = lane >> 4;
  const int m = lane & 15;
  const int rb = blockIdx.x * 16;
  const int cb = blockIdx.y * 256 + wave * 64;
  const int Npad = (Nv + 15) & ~15;
  if (cb >= Npad) return;  // uniform per wave

  FragAcc acc[4];
#pragma unroll
  for (int t = 0; t < 4; ++t)
#pragma unroll
    for (int r = 0; r < 8; ++r) acc[t].f[r] = 0.0f;

  const float* ap = A + (size_t)(rb + m) * KK;
  const float* wp[4];
#pragma unroll
  for (int t = 0; t < 4; ++t) {
    int n = cb + t * 16 + m;
    n = n < Nv ? n : Nv - 1;           // clamp tail reads
    wp[t] = W + (size_t)n * KK + (g << 4);
  }
#pragma unroll
  for (int kt = 0; kt < (KK >> 5); ++kt) {
    // batch all fragment loads (A + 4xB) before the wmma burst
    float2 av[8];
#pragma unroll
    for (int v = 0; v < 8; ++v) {
      int k0 = kt * 32 + (g << 4) + (v << 1);
      av[v] = *(const float2*)(ap + k0);
    }
    float2 wv[4][8];
#pragma unroll
    for (int t = 0; t < 4; ++t)
#pragma unroll
      for (int v = 0; v < 8; ++v) wv[t][v] = *(const float2*)(wp[t] + kt * 32 + 2 * v);
    FragB a;
#pragma unroll
    for (int v = 0; v < 8; ++v) a.u[v] = pack2(av[v].x, av[v].y);
#pragma unroll
    for (int t = 0; t < 4; ++t) {
      FragB b;
#pragma unroll
      for (int v = 0; v < 8; ++v) b.u[v] = pack2(wv[t][v].x, wv[t][v].y);
      acc[t].v = __builtin_amdgcn_wmma_f32_16x16x32_bf16(
          false, a.v, false, b.v, (short)0, acc[t].v, false, false);
    }
  }
#pragma unroll
  for (int t = 0; t < 4; ++t) {
    int n = cb + t * 16 + m;
    if (n < Nv) {
      float bs = bias ? bias[n] : 0.0f;
#pragma unroll
      for (int r = 0; r < 8; ++r) {
        float v = acc[t].f[r] + bs;
        if (relu) v = fmaxf(v, 0.0f);
        C[(size_t)(rb + r + 8 * g) * ldc + n] = v;
      }
    }
  }
}

// ---------------------------------------------------------------------------
// Small elementwise / scatter kernels
// ---------------------------------------------------------------------------
__global__ __launch_bounds__(256) void k_zero(float* __restrict__ p, int n) {
  int i = blockIdx.x * 256 + threadIdx.x;
  if (i < n) p[i] = 0.0f;
}
__global__ __launch_bounds__(256) void k_counts(const int* __restrict__ edge_index,
                                                float* __restrict__ counts) {
  int e = blockIdx.x * 256 + threadIdx.x;
  if (e < EE) atomicAdd(counts + edge_index[EE + e], 1.0f);
}
__global__ __launch_bounds__(256) void k_scatter(const float* __restrict__ msg,
                                                 const int* __restrict__ edge_index,
                                                 float* __restrict__ agg) {
  int tid = blockIdx.x * 256 + threadIdx.x;   // E*256
  int e = tid >> 8, o = tid & 255;
  atomicAdd(agg + (size_t)edge_index[EE + e] * HH + o, msg[tid]);
}
__global__ __launch_bounds__(256) void k_mgate(const float* __restrict__ agg,
                                               const float* __restrict__ counts,
                                               const float* __restrict__ conv_b,
                                               float* __restrict__ mbuf) {
  int tid = blockIdx.x * 256 + threadIdx.x;   // N*256
  float c = counts[tid >> 8];
  c = c < 1.0f ? 1.0f : c;
  mbuf[tid] = fmaxf(agg[tid] / c + conv_b[tid & 255], 0.0f);
}
__global__ __launch_bounds__(256) void k_gru(const float* __restrict__ gi,
                                             const float* __restrict__ gh,
                                             const float* __restrict__ h_old,
                                             float* __restrict__ h_new,
                                             unsigned short* __restrict__ out_bf) {
  int tid = blockIdx.x * 256 + threadIdx.x;   // N*256
  int n = tid >> 8, c = tid & 255;
  size_t base = (size_t)n * 768 + c;
  float r  = sigmf(gi[base] + gh[base]);
  float z  = sigmf(gi[base + 256] + gh[base + 256]);
  float nn = tanhf(gi[base + 512] + r * gh[base + 512]);
  float hv = h_old[tid];
  float hn = (1.0f - z) * nn + z * hv;
  h_new[tid] = hn;
  out_bf[tid] = f2bf(hn);
}
__global__ __launch_bounds__(256) void k_cat(const float* __restrict__ h,
                                             const int* __restrict__ edge_index,
                                             float* __restrict__ cat) {
  int tid = blockIdx.x * 256 + threadIdx.x;   // E*512
  int e = tid >> 9, c = tid & 511;
  int node = (c < 256) ? edge_index[e] : edge_index[EE + e];
  cat[tid] = h[(size_t)node * HH + (c & 255)];
}
__global__ __launch_bounds__(256) void k_bnstats(const float* __restrict__ h1,
                                                 float* __restrict__ mu,
                                                 float* __restrict__ var) {
  __shared__ float s1[256], s2[256];
  int c = blockIdx.x, t = threadIdx.x;
  float a = 0.0f, b = 0.0f;
  for (int e = t; e < EE; e += 256) {
    float v = h1[(size_t)e * 128 + c];
    a += v; b += v * v;
  }
  s1[t] = a; s2[t] = b;
  __syncthreads();
  for (int st = 128; st > 0; st >>= 1) {
    if (t < st) { s1[t] += s1[t + st]; s2[t] += s2[t + st]; }
    __syncthreads();
  }
  if (t == 0) {
    float mm = s1[0] / (float)EE;
    mu[c] = mm;
    var[c] = s2[0] / (float)EE - mm * mm;
  }
}
__global__ __launch_bounds__(256) void k_precs(const float* __restrict__ h1,
                                               const float* __restrict__ mu,
                                               const float* __restrict__ var,
                                               const float* __restrict__ g,
                                               const float* __restrict__ b,
                                               const float* __restrict__ W2,
                                               const float* __restrict__ b2,
                                               float* __restrict__ out) {
  int e = blockIdx.x * 256 + threadIdx.x;
  float acc = b2[0];
#pragma unroll 4
  for (int c = 0; c < 128; ++c) {
    float xn = (h1[(size_t)e * 128 + c] - mu[c]) * rsqrtf(var[c] + 1e-5f) * g[c] + b[c];
    acc += fmaxf(xn, 0.0f) * W2[c];
  }
  out[e] = acc;
}

// ---------------------------------------------------------------------------
extern "C" void kernel_launch(void* const* d_in, const int* in_sizes, int n_in,
                              void* d_out, int out_size, void* d_ws, size_t ws_size,
                              hipStream_t stream) {
  const int*   x        = (const int*)  d_in[0];
  const float* pos      = (const float*)d_in[1];
  const int*   eidx     = (const int*)  d_in[2];
  const float* eatr     = (const float*)d_in[3];
  const float* emb      = (const float*)d_in[4];
  const float* lin0_W   = (const float*)d_in[5];
  const float* lin0_b   = (const float*)d_in[6];
  const float* mlp1_W   = (const float*)d_in[7];
  const float* mlp1_b   = (const float*)d_in[8];
  const float* mlp2_W   = (const float*)d_in[9];
  const float* conv_b   = (const float*)d_in[11];
  const float* gru_Wih  = (const float*)d_in[12];
  const float* gru_Whh  = (const float*)d_in[13];
  const float* gru_bih  = (const float*)d_in[14];
  const float* gru_bhh  = (const float*)d_in[15];
  const float* lin1_W   = (const float*)d_in[16];
  const float* lin1_b   = (const float*)d_in[17];
  const float* lin2_W   = (const float*)d_in[18];
  const float* lin2_b   = (const float*)d_in[19];
  const float* l6_W1    = (const float*)d_in[20];
  const float* l6_b1    = (const float*)d_in[21];
  const float* bn_g     = (const float*)d_in[22];
  const float* bn_b     = (const float*)d_in[23];
  const float* l6_W2    = (const float*)d_in[24];
  const float* l6_b2    = (const float*)d_in[25];
  float* out = (float*)d_out;

  // Workspace carve (~60 MB)
  size_t off = 0;
  auto carve = [&](size_t bytes) {
    void* p = (char*)d_ws + off;
    off = (off + bytes + 255) & ~(size_t)255;
    return p;
  };
  float*          hA     = (float*)         carve((size_t)NN * HH * 4);
  float*          hB     = (float*)         carve((size_t)NN * HH * 4);
  unsigned short* out_bf = (unsigned short*)carve((size_t)NN * HH * 2);
  unsigned short* he_bf  = (unsigned short*)carve((size_t)EE * 128 * 2);
  // +1 ktile of padding (16 KB) so the software pipeline may over-issue one tile
  unsigned int*   Bp     = (unsigned int*)  carve((size_t)(KTILES + 1) * BSTRIDE);
  float*          msg    = (float*)         carve((size_t)EE * HH * 4);
  float*          agg    = (float*)         carve((size_t)NN * HH * 4);
  float*          counts = (float*)         carve((size_t)NN * 4);
  float*          mbuf   = (float*)         carve((size_t)NN * HH * 4);
  float*          gi     = (float*)         carve((size_t)NN * 768 * 4);
  float*          gh     = (float*)         carve((size_t)NN * 768 * 4);
  float*          cat    = (float*)         carve((size_t)EE * 512 * 4);
  float*          o1     = (float*)         carve((size_t)EE * HH * 4);
  float*          h1     = (float*)         carve((size_t)EE * 128 * 4);
  float*          mu     = (float*)         carve(128 * 4);
  float*          var    = (float*)         carve(128 * 4);
  (void)ws_size; (void)in_sizes; (void)n_in; (void)out_size;

  // Pre-pass
  k_pack_B<<<(KBIG * HH / 2) / 256, 256, 0, stream>>>(mlp2_W, Bp);
  k_embed_lin0<<<NN, 256, 0, stream>>>(x, pos, emb, lin0_W, lin0_b, hA, out_bf);
  k_he<<<EE * 128 / 256, 256, 0, stream>>>(eatr, mlp1_W, mlp1_b, he_bf);
  k_zero<<<(NN + 255) / 256, 256, 0, stream>>>(counts, NN);
  k_counts<<<EE / 256, 256, 0, stream>>>(eidx, counts);

  // 3 message-passing iterations (h ping-pongs hA -> hB -> hA -> hB)
  float* hcur = hA;
  float* hnxt = hB;
  for (int it = 0; it < 3; ++it) {
    k_msg<<<EE / 32, 256, 0, stream>>>(out_bf, he_bf, Bp, eidx, msg);
    k_zero<<<(NN * HH) / 256, 256, 0, stream>>>(agg, NN * HH);
    k_scatter<<<(EE * HH) / 256, 256, 0, stream>>>(msg, eidx, agg);
    k_mgate<<<(NN * HH) / 256, 256, 0, stream>>>(agg, counts, conv_b, mbuf);
    // gi = m @ Wih^T + bih ; gh = h @ Whh^T + bhh   (M=2000, K=256, N=768)
    k_gemm<256><<<dim3(NN / 16, 3), 128, 0, stream>>>(mbuf, gru_Wih, gru_bih, gi, 768, 768, 0);
    k_gemm<256><<<dim3(NN / 16, 3), 128, 0, stream>>>(hcur, gru_Whh, gru_bhh, gh, 768, 768, 0);
    k_gru<<<(NN * HH) / 256, 256, 0, stream>>>(gi, gh, hcur, hnxt, out_bf);
    float* t = hcur; hcur = hnxt; hnxt = t;
  }

  // Heads
  k_cat<<<(EE * 512) / 256, 256, 0, stream>>>(hcur, eidx, cat);
  k_gemm<512><<<dim3(EE / 16, 1), 128, 0, stream>>>(cat, lin1_W, lin1_b, o1, 256, 256, 1);
  k_gemm<256><<<dim3(EE / 16, 1), 128, 0, stream>>>(o1, lin2_W, lin2_b, out, 242, 242, 0);
  k_gemm<512><<<dim3(EE / 16, 1), 128, 0, stream>>>(cat, l6_W1, l6_b1, h1, 128, 128, 0);
  k_bnstats<<<128, 256, 0, stream>>>(h1, mu, var);
  k_precs<<<EE / 256, 256, 0, stream>>>(h1, mu, var, bn_g, bn_b, l6_W2, l6_b2,
                                        out + (size_t)EE * 242);
}
